// SparseAttention_35424890258072
// MI455X (gfx1250) — matrix operand order repaired and checked by reading, not compiled, not run on hardware
//
#include <hip/hip_runtime.h>
#include <hip/hip_bf16.h>

typedef __attribute__((ext_vector_type(16))) _Float16 v16h;
typedef __attribute__((ext_vector_type(8)))  _Float16 v8h;
typedef __attribute__((ext_vector_type(8)))  float    v8f;

#define S_LEN 2048
#define DM    512
#define HEADS 8
#define HD    64
#define BAND  512
#define NEGC  (-1.0e9f)

static __device__ __forceinline__ v8f wmma_f16(v16h a, v16h b, v8f c) {
  // D = A(16x32 f16) * B(32x16 f16) + C(16x16 f32)
  return __builtin_amdgcn_wmma_f32_16x16x32_f16(
      /*neg_a=*/false, a, /*neg_b=*/false, b,
      /*c_mod=*/(short)0, c, /*reuse_a=*/false, /*reuse_b=*/false);
}

// ---------------------------------------------------------------------------
// One-shot f32 -> f16 conversion of a weight matrix (done once per weight so
// the GEMM hot loops issue pure v16h B-fragment loads with no cvt VALU).
// 8 elements per thread, vectorized.
// ---------------------------------------------------------------------------
__global__ __launch_bounds__(256) void cvt_f16(const float* __restrict__ src,
                                               _Float16* __restrict__ dst, int n8) {
  const int i = blockIdx.x * blockDim.x + threadIdx.x;
  if (i < n8) {
    const float4* p = (const float4*)src + (size_t)i * 2;
    const float4 x0 = p[0], x1 = p[1];
    v8h o;
    o[0] = (_Float16)x0.x; o[1] = (_Float16)x0.y; o[2] = (_Float16)x0.z; o[3] = (_Float16)x0.w;
    o[4] = (_Float16)x1.x; o[5] = (_Float16)x1.y; o[6] = (_Float16)x1.z; o[7] = (_Float16)x1.w;
    *(v8h*)(dst + (size_t)i * 8) = o;
  }
}

// ---------------------------------------------------------------------------
// Projection GEMM: Y[m,n] = sum_k X[m,k] * W[n,k] + bias[n]
// X f32 [8192, 512] row-major, Wh f16 [512,512] row-major (pre-converted).
// MODE 0: store f16 head-split row-major  [B,H,S,hd]   (Q, K)
// MODE 1: store f16 head-split transposed [B,H,hd,S]   (V)
// Block: 256 threads = 8 waves; wave tile 32(M) x 64(N); WG tile 256 x 64.
// ---------------------------------------------------------------------------
template<int MODE>
__global__ __launch_bounds__(256) void proj_gemm(
    const float* __restrict__ X, const _Float16* __restrict__ Wh,
    const float* __restrict__ bias, _Float16* __restrict__ Y)
{
  const int lane  = threadIdx.x & 31;
  const int wv    = threadIdx.x >> 5;
  const int lr    = lane & 15;
  const int hi    = lane >> 4;
  const int mBase = blockIdx.x * 256 + wv * 32;
  const int nBase = blockIdx.y * 64;

  v8f acc[2][4];
  #pragma unroll
  for (int mi = 0; mi < 2; ++mi)
    #pragma unroll
    for (int nt = 0; nt < 4; ++nt) acc[mi][nt] = (v8f){};

  for (int kt = 0; kt < DM / 32; ++kt) {
    const int kBase = kt * 32;
    // A fragments: 16x32, lane<16: K {0-15}, lane>=16: K {16-31}
    v16h a[2];
    #pragma unroll
    for (int mi = 0; mi < 2; ++mi) {
      const float* p0 = X + (size_t)(mBase + mi * 16 + lr) * DM + kBase + hi * 16;
      const float* p1 = p0 + 8;
      #pragma unroll
      for (int i = 0; i < 8; ++i) { a[mi][i] = (_Float16)p0[i]; a[mi][8 + i] = (_Float16)p1[i]; }
    }
    // B fragments: 32x16, lane column n = W row (n), 32B contiguous f16 loads
    #pragma unroll
    for (int nt = 0; nt < 4; ++nt) {
      const v16h b = *(const v16h*)(Wh + (size_t)(nBase + nt * 16 + lr) * DM + kBase + hi * 16);
      acc[0][nt] = wmma_f16(a[0], b, acc[0][nt]);
      acc[1][nt] = wmma_f16(a[1], b, acc[1][nt]);
    }
  }

  #pragma unroll
  for (int nt = 0; nt < 4; ++nt) {
    const int n  = nBase + nt * 16 + lr;
    const float bval = bias[n];
    const int h = n >> 6, d = n & 63;
    #pragma unroll
    for (int mi = 0; mi < 2; ++mi) {
      #pragma unroll
      for (int r = 0; r < 8; ++r) {
        const int row = mBase + mi * 16 + hi * 8 + r;
        const int bb = row >> 11, s = row & 2047;
        const float val = acc[mi][nt][r] + bval;
        if (MODE == 0)
          Y[(((size_t)(bb * HEADS + h) * S_LEN + s) << 6) + d] = (_Float16)val;
        else
          Y[(((size_t)(bb * HEADS + h) * HD + d) << 11) + s] = (_Float16)val;
      }
    }
  }
}

// ---------------------------------------------------------------------------
// Output projection: out[m,n] = sum_k ctx_h[m,k] * Wo[n,k] + bo[n], f32 store.
// ctx_h is f16 stored as [B,H,S,hd]; logical column k = h*64+d. Wo pre-f16.
// ---------------------------------------------------------------------------
__global__ __launch_bounds__(256) void oproj_gemm(
    const _Float16* __restrict__ Ch, const _Float16* __restrict__ Wh,
    const float* __restrict__ bias, float* __restrict__ out)
{
  const int lane  = threadIdx.x & 31;
  const int wv    = threadIdx.x >> 5;
  const int lr    = lane & 15;
  const int hi    = lane >> 4;
  const int mBase = blockIdx.x * 256 + wv * 32;
  const int nBase = blockIdx.y * 64;

  v8f acc[2][4];
  #pragma unroll
  for (int mi = 0; mi < 2; ++mi)
    #pragma unroll
    for (int nt = 0; nt < 4; ++nt) acc[mi][nt] = (v8f){};

  for (int kt = 0; kt < DM / 32; ++kt) {
    const int kBase = kt * 32;
    const int kc0 = kBase + hi * 16;   // multiple of 8, stays inside one head
    const int kc1 = kc0 + 8;
    v16h a[2];
    #pragma unroll
    for (int mi = 0; mi < 2; ++mi) {
      const int row = mBase + mi * 16 + lr;
      const int bb = row >> 11, s = row & 2047;
      const v8h lo = *(const v8h*)(Ch + (((size_t)(bb * HEADS + (kc0 >> 6)) * S_LEN + s) << 6) + (kc0 & 63));
      const v8h hh = *(const v8h*)(Ch + (((size_t)(bb * HEADS + (kc1 >> 6)) * S_LEN + s) << 6) + (kc1 & 63));
      #pragma unroll
      for (int i = 0; i < 8; ++i) { a[mi][i] = lo[i]; a[mi][8 + i] = hh[i]; }
    }
    #pragma unroll
    for (int nt = 0; nt < 4; ++nt) {
      const v16h b = *(const v16h*)(Wh + (size_t)(nBase + nt * 16 + lr) * DM + kBase + hi * 16);
      acc[0][nt] = wmma_f16(a[0], b, acc[0][nt]);
      acc[1][nt] = wmma_f16(a[1], b, acc[1][nt]);
    }
  }

  #pragma unroll
  for (int nt = 0; nt < 4; ++nt) {
    const int n = nBase + nt * 16 + lr;
    const float bval = bias[n];
    #pragma unroll
    for (int mi = 0; mi < 2; ++mi) {
      #pragma unroll
      for (int r = 0; r < 8; ++r) {
        const int row = mBase + mi * 16 + hi * 8 + r;
        out[(size_t)row * DM + n] = acc[mi][nt][r] + bval;
      }
    }
  }
}

// ---------------------------------------------------------------------------
// Banded attention. Block = 128 threads = 4 waves; each wave owns 16 queries.
// Pass 1: online row max/sum over the band (QK^T via WMMA).
// Pass 2: recompute scores, write normalized probs (zeros out of band) to
//         attn, route P through LDS (C-layout -> A-layout) and accumulate
//         P@V via WMMA into f32 context, stored as f16 for the O-projection.
// ---------------------------------------------------------------------------
__global__ __launch_bounds__(128) void band_attn(
    const _Float16* __restrict__ Qh, const _Float16* __restrict__ Kh,
    const _Float16* __restrict__ Vt, _Float16* __restrict__ Ch,
    float* __restrict__ attn)
{
  __shared__ __align__(16) _Float16 Pbuf[4][16][40]; // 40-half row pitch: 16B-aligned rows
  const int lane  = threadIdx.x & 31;
  const int wv    = threadIdx.x >> 5;
  const int lr    = lane & 15;
  const int hi    = lane >> 4;
  const int bh    = blockIdx.x >> 5;                    // 0..31
  const int qBase = ((blockIdx.x & 31) << 6) + (wv << 4);

  const _Float16* Qb = Qh + ((size_t)bh << 17);  // 2048*64
  const _Float16* Kb = Kh + ((size_t)bh << 17);
  const _Float16* Vb = Vt + ((size_t)bh << 17);
  float* Ab = attn + ((size_t)bh << 22);         // 2048*2048

  // Q A-fragments for hd 0-31 (a0) and 32-63 (a1)
  v16h a0, a1;
  {
    const _Float16* qp = Qb + ((size_t)(qBase + lr) << 6) + (hi << 4);
    v8h x0 = *(const v8h*)(qp);
    v8h x1 = *(const v8h*)(qp + 8);
    v8h x2 = *(const v8h*)(qp + 32);
    v8h x3 = *(const v8h*)(qp + 40);
    #pragma unroll
    for (int i = 0; i < 8; ++i) { a0[i] = x0[i]; a0[8 + i] = x1[i]; a1[i] = x2[i]; a1[8 + i] = x3[i]; }
  }

  const int lo0 = qBase - BAND;
  const int jt0 = (lo0 > 0) ? (lo0 >> 4) : 0;
  int jt1 = (qBase + BAND + 15) >> 4;
  if (jt1 > S_LEN / 16 - 1) jt1 = S_LEN / 16 - 1;
  const float scale = 0.125f; // 1/sqrt(64)
  const int rbase = qBase + (hi << 3);

  float m[8], l[8];
  #pragma unroll
  for (int r = 0; r < 8; ++r) { m[r] = -3.0e38f; l[r] = 0.0f; }

  // ---- Pass 1: row statistics over the band ----
  for (int jt = jt0; jt <= jt1; ++jt) {
    const int keyBase = jt << 4;
    const _Float16* kp = Kb + ((size_t)(keyBase + lr) << 6) + (hi << 4);
    __builtin_prefetch((const void*)(kp + 1024), 0, 0); // next 16-key tile -> global_prefetch_b8
    v16h b0 = *(const v16h*)(kp);
    v16h b1 = *(const v16h*)(kp + 32);
    v8f s = (v8f){};
    s = wmma_f16(a0, b0, s);
    s = wmma_f16(a1, b1, s);
    const int col = keyBase + lr;
    float sv[8];
    #pragma unroll
    for (int r = 0; r < 8; ++r) {
      const int row = rbase + r;
      float v = s[r] * scale;
      const int dd = row - col;
      if (dd > BAND || dd < -BAND) v = NEGC;
      sv[r] = v;
    }
    #pragma unroll
    for (int r = 0; r < 8; ++r) {
      float v = sv[r];
      v = fmaxf(v, __shfl_xor(v, 1, 32));
      v = fmaxf(v, __shfl_xor(v, 2, 32));
      v = fmaxf(v, __shfl_xor(v, 4, 32));
      v = fmaxf(v, __shfl_xor(v, 8, 32));
      const float mn = fmaxf(m[r], v);
      float e = __expf(sv[r] - mn);
      e += __shfl_xor(e, 1, 32);
      e += __shfl_xor(e, 2, 32);
      e += __shfl_xor(e, 4, 32);
      e += __shfl_xor(e, 8, 32);
      l[r] = l[r] * __expf(m[r] - mn) + e;
      m[r] = mn;
    }
  }
  float linv[8];
  #pragma unroll
  for (int r = 0; r < 8; ++r) linv[r] = 1.0f / l[r];

  // ---- Pass 2: write attn (full rows) + accumulate P@V ----
  v8f c0 = (v8f){}, c1 = (v8f){}, c2 = (v8f){}, c3 = (v8f){};
  for (int jt2 = 0; jt2 < S_LEN / 16; jt2 += 2) {
    const bool anyband = (jt2 + 1 >= jt0) && (jt2 <= jt1);
    #pragma unroll
    for (int t = 0; t < 2; ++t) {
      const int jt = jt2 + t;
      const int keyBase = jt << 4;
      const int col = keyBase + lr;
      float p[8];
      if (jt >= jt0 && jt <= jt1) {
        const _Float16* kp = Kb + ((size_t)(keyBase + lr) << 6) + (hi << 4);
        v16h b0 = *(const v16h*)(kp);
        v16h b1 = *(const v16h*)(kp + 32);
        v8f s = (v8f){};
        s = wmma_f16(a0, b0, s);
        s = wmma_f16(a1, b1, s);
        #pragma unroll
        for (int r = 0; r < 8; ++r) {
          const int row = rbase + r;
          float v = s[r] * scale;
          const int dd = row - col;
          if (dd > BAND || dd < -BAND) v = NEGC;
          p[r] = __expf(v - m[r]) * linv[r];   // masked -> exp(-1e9-m) == 0
        }
      } else {
        #pragma unroll
        for (int r = 0; r < 8; ++r) p[r] = 0.0f;
      }
      float* ap = Ab + ((size_t)rbase << 11) + col;
      #pragma unroll
      for (int r = 0; r < 8; ++r) ap[(size_t)r << 11] = p[r];
      if (anyband) {
        #pragma unroll
        for (int r = 0; r < 8; ++r)
          Pbuf[wv][(hi << 3) + r][(t << 4) + lr] = (_Float16)p[r];
      }
    }
    if (anyband) {
      asm volatile("s_wait_dscnt 0" ::: "memory"); // cross-lane LDS exchange (intra-wave)
      v16h aP;
      {
        const _Float16* pp = &Pbuf[wv][lr][hi << 4];
        v8h plo = *(const v8h*)(pp);
        v8h phi = *(const v8h*)(pp + 8);
        #pragma unroll
        for (int i = 0; i < 8; ++i) { aP[i] = plo[i]; aP[8 + i] = phi[i]; }
      }
      const int keyB = jt2 << 4;
      const _Float16* vp = Vb + ((size_t)lr << 11) + keyB + (hi << 4);
      v16h bv0 = *(const v16h*)(vp);
      v16h bv1 = *(const v16h*)(vp + ((size_t)16 << 11));
      v16h bv2 = *(const v16h*)(vp + ((size_t)32 << 11));
      v16h bv3 = *(const v16h*)(vp + ((size_t)48 << 11));
      c0 = wmma_f16(aP, bv0, c0);
      c1 = wmma_f16(aP, bv1, c1);
      c2 = wmma_f16(aP, bv2, c2);
      c3 = wmma_f16(aP, bv3, c3);
    }
  }

  // store context as f16 [B,H,S,hd]
  _Float16* cp = Ch + ((size_t)bh << 17) + ((size_t)rbase << 6) + lr;
  v8f cc[4] = {c0, c1, c2, c3};
  #pragma unroll
  for (int t = 0; t < 4; ++t)
    #pragma unroll
    for (int r = 0; r < 8; ++r)
      cp[((size_t)r << 6) + (t << 4)] = (_Float16)cc[t][r];
}

// ---------------------------------------------------------------------------
extern "C" void kernel_launch(void* const* d_in, const int* in_sizes, int n_in,
                              void* d_out, int out_size, void* d_ws, size_t ws_size,
                              hipStream_t stream) {
  (void)in_sizes; (void)n_in; (void)out_size; (void)ws_size;
  const float* q  = (const float*)d_in[0];
  const float* kx = (const float*)d_in[1];
  const float* vx = (const float*)d_in[2];
  const float* wq = (const float*)d_in[3];
  const float* bq = (const float*)d_in[4];
  const float* wk = (const float*)d_in[5];
  const float* bk = (const float*)d_in[6];
  const float* wvw= (const float*)d_in[7];
  const float* bv = (const float*)d_in[8];
  const float* wo = (const float*)d_in[9];
  const float* bo = (const float*)d_in[10];

  const size_t PROJ = (size_t)4 * S_LEN * DM;  // 4,194,304 elements
  const size_t WSZ  = (size_t)DM * DM;         //   262,144 elements
  _Float16* Qh  = (_Float16*)d_ws;
  _Float16* Kh  = Qh + PROJ;
  _Float16* Vt  = Kh + PROJ;
  _Float16* Ch  = Vt + PROJ;
  _Float16* Wqh = Ch + PROJ;
  _Float16* Wkh = Wqh + WSZ;
  _Float16* Wvh = Wkh + WSZ;
  _Float16* Woh = Wvh + WSZ;
  float* out  = (float*)d_out;
  float* attn = out + PROJ;

  // Pre-convert the four weight matrices to f16 (once, tiny).
  const int n8 = (int)(WSZ / 8);               // 32768 threads of 8 elems
  dim3 cb(256), cg((n8 + 255) / 256);
  cvt_f16<<<cg, cb, 0, stream>>>(wq,  Wqh, n8);
  cvt_f16<<<cg, cb, 0, stream>>>(wk,  Wkh, n8);
  cvt_f16<<<cg, cb, 0, stream>>>(wvw, Wvh, n8);
  cvt_f16<<<cg, cb, 0, stream>>>(wo,  Woh, n8);

  dim3 pb(256), pg(8192 / 256, 512 / 64);  // 32 x 8 workgroups
  proj_gemm<0><<<pg, pb, 0, stream>>>(q,  Wqh, bq, Qh);
  proj_gemm<0><<<pg, pb, 0, stream>>>(kx, Wkh, bk, Kh);
  proj_gemm<1><<<pg, pb, 0, stream>>>(vx, Wvh, bv, Vt);
  band_attn<<<dim3(1024), dim3(128), 0, stream>>>(Qh, Kh, Vt, Ch, attn);
  oproj_gemm<<<pg, pb, 0, stream>>>(Ch, Woh, bo, out);
}
